// SparseCausalSelfAttention_3487513444756
// MI455X (gfx1250) — matrix operand order repaired and checked by reading, not compiled, not run on hardware
//
#include <hip/hip_runtime.h>
#include <hip/hip_bf16.h>

#define DEVFN __device__ __forceinline__

typedef __attribute__((ext_vector_type(16))) __bf16 bf16x16;
typedef __attribute__((ext_vector_type(8)))  float  f32x8;

namespace {
constexpr int BB = 2, SS = 2048, EE = 1024, HH = 16, DD = 64;
constexpr int E3 = 3 * EE;           // 3072
constexpr int MM = BB * SS;          // 4096
}

union FragU { bf16x16 v; uint4 u[2]; };

// A fragment (16x32 bf16, M x K), row-major source with leading dim ld.
// lane<16: row=lane, slots K = {k0..k0+7, k0+16..k0+23}
// lane>=16: row=lane-16, slots K = {k0+8..k0+15, k0+24..k0+31}
DEVFN bf16x16 load_a_frag(const __bf16* base, int ld, int lane) {
  const int half = lane >> 4;
  const int row  = lane & 15;
  const __bf16* p = base + (size_t)row * ld + half * 8;
  FragU f;
  f.u[0] = *(const uint4*)(p);
  f.u[1] = *(const uint4*)(p + 16);   // +16 elements = +32 bytes
  return f.v;
}

// B fragment (32x16 bf16, K x N), K-major source with leading dim ld.
// lane L holds column N=L%16, K = {8h..8h+7, 16+8h..16+8h+7}, h=L/16.
DEVFN bf16x16 load_b_frag(const __bf16* base, int ld, int lane) {
  const __bf16* p = base + (lane & 15) + (size_t)((lane >> 4) * 8) * ld;
  bf16x16 v;
#pragma unroll
  for (int s = 0; s < 8; s++) {
    v[s]     = p[(size_t)s * ld];
    v[8 + s] = p[(size_t)(16 + s) * ld];
  }
  return v;
}

DEVFN f32x8 wmma_bf16(bf16x16 a, bf16x16 b, f32x8 c) {
  return __builtin_amdgcn_wmma_f32_16x16x32_bf16(
      /*neg_a=*/false, a, /*neg_b=*/false, b,
      /*c_mod=*/(short)0, c, /*reuse_a=*/false, /*reuse_b=*/false);
}

// ---------------------------------------------------------------------------
// Conversion / layout kernels
// ---------------------------------------------------------------------------
__global__ void cvt_f32_bf16(const float* __restrict__ in, __bf16* __restrict__ out, int n) {
  int i = blockIdx.x * blockDim.x + threadIdx.x;
  if (i < n) out[i] = (__bf16)in[i];
}

// in: [N,K] row-major fp32  ->  out: [K,N] row-major bf16
__global__ void cvt_transpose(const float* __restrict__ in, __bf16* __restrict__ out,
                              int N, int K) {
  int i = blockIdx.x * blockDim.x + threadIdx.x;
  if (i < N * K) {
    int k = i / N, n = i - k * N;
    out[i] = (__bf16)in[(size_t)n * K + k];
  }
}

// kt[((b*H+h)*D + d)*S + s] = qkv[(b*S+s)*3E + E + h*D + d]
__global__ void build_kt(const __bf16* __restrict__ qkv, __bf16* __restrict__ kt) {
  int i = blockIdx.x * blockDim.x + threadIdx.x;   // over B*H*D*S
  if (i < BB * HH * DD * SS) {
    int s = i & (SS - 1);
    int t = i >> 11;                 // / S
    int d = t & (DD - 1);
    t >>= 6;                         // / D
    int h = t & (HH - 1);
    int b = t >> 4;                  // / H
    kt[i] = qkv[((size_t)(b * SS + s)) * E3 + EE + h * DD + d];
  }
}

// ---------------------------------------------------------------------------
// GEMM: C[M,N] = A[M,K] (row-major bf16) x Bk[K,N] (K-major bf16) + bias[N]
// block = 256 threads = 8 waves; block tile 128(M) x 128(N); wave tile 32x64.
// ---------------------------------------------------------------------------
template <typename OutT>
__global__ __launch_bounds__(256) void gemm_nt(const __bf16* __restrict__ A,
                                               const __bf16* __restrict__ Bk,
                                               const float* __restrict__ bias,
                                               OutT* __restrict__ C, int N, int K) {
  const int lane = threadIdx.x & 31;
  const int wv   = threadIdx.x >> 5;
  const int wm   = wv & 3, wn = wv >> 2;
  const int m0   = blockIdx.y * 128 + wm * 32;
  const int n0   = blockIdx.x * 128 + wn * 64;

  const f32x8 fzero = {0.f, 0.f, 0.f, 0.f, 0.f, 0.f, 0.f, 0.f};
  f32x8 acc[2][4];
#pragma unroll
  for (int i = 0; i < 2; i++)
#pragma unroll
    for (int j = 0; j < 4; j++) acc[i][j] = fzero;

  for (int k0 = 0; k0 < K; k0 += 32) {
    bf16x16 a0 = load_a_frag(A + (size_t)m0 * K + k0, K, lane);
    bf16x16 a1 = load_a_frag(A + (size_t)(m0 + 16) * K + k0, K, lane);
    bf16x16 b0 = load_b_frag(Bk + (size_t)k0 * N + n0 +  0, N, lane);
    bf16x16 b1 = load_b_frag(Bk + (size_t)k0 * N + n0 + 16, N, lane);
    bf16x16 b2 = load_b_frag(Bk + (size_t)k0 * N + n0 + 32, N, lane);
    bf16x16 b3 = load_b_frag(Bk + (size_t)k0 * N + n0 + 48, N, lane);
    acc[0][0] = wmma_bf16(a0, b0, acc[0][0]);
    acc[0][1] = wmma_bf16(a0, b1, acc[0][1]);
    acc[0][2] = wmma_bf16(a0, b2, acc[0][2]);
    acc[0][3] = wmma_bf16(a0, b3, acc[0][3]);
    acc[1][0] = wmma_bf16(a1, b0, acc[1][0]);
    acc[1][1] = wmma_bf16(a1, b1, acc[1][1]);
    acc[1][2] = wmma_bf16(a1, b2, acc[1][2]);
    acc[1][3] = wmma_bf16(a1, b3, acc[1][3]);
  }

  const int half = lane >> 4, c0 = lane & 15;
#pragma unroll
  for (int mi = 0; mi < 2; mi++)
#pragma unroll
    for (int nj = 0; nj < 4; nj++)
#pragma unroll
      for (int r = 0; r < 8; r++) {
        int row = m0 + mi * 16 + half * 8 + r;
        int col = n0 + nj * 16 + c0;
        float v = acc[mi][nj][r] + bias[col];
        C[(size_t)row * N + col] = (OutT)v;
      }
}

// ---------------------------------------------------------------------------
// Flash attention: block = (b, h, 64 q rows), 4 waves x 16 q rows each.
// qkv: bf16 [B,S,3E]; kt: bf16 [B,H,D,S]; ctx out: bf16 [B,S,E]
// ---------------------------------------------------------------------------
__global__ __launch_bounds__(128) void attn_kernel(const __bf16* __restrict__ qkv,
                                                   const __bf16* __restrict__ kt,
                                                   __bf16* __restrict__ ctx) {
  __shared__ __bf16 plds[4 * 16 * 32];
  const int lane = threadIdx.x & 31;
  const int wv   = threadIdx.x >> 5;
  const int qb   = blockIdx.x * 64 + wv * 16;
  const int h    = blockIdx.y, b = blockIdx.z;
  __bf16* pw = plds + wv * (16 * 32);

  // Q fragments for this wave's 16 rows (d = 64 -> two 32-wide A frags)
  const __bf16* qbase = qkv + ((size_t)(b * SS + qb)) * E3 + h * DD;
  bf16x16 qa0 = load_a_frag(qbase + 0, E3, lane);
  bf16x16 qa1 = load_a_frag(qbase + 32, E3, lane);

  const f32x8 fzero = {0.f, 0.f, 0.f, 0.f, 0.f, 0.f, 0.f, 0.f};
  f32x8 o[4] = {fzero, fzero, fzero, fzero};
  float rmax[8], rsum[8];
#pragma unroll
  for (int r = 0; r < 8; r++) { rmax[r] = -1e30f; rsum[r] = 0.f; }

  const int half = lane >> 4, c0 = lane & 15;
  const int rbase = qb + half * 8;

  const __bf16* ktbase = kt + ((size_t)(b * HH + h)) * DD * SS;
  const __bf16* vbase  = qkv + ((size_t)b * SS) * E3 + 2 * EE + h * DD;

  for (int kv0 = 0; kv0 < qb + 16; kv0 += 32) {
    // S = Q @ K^T over d=64: two n-frags (16 keys each), two k-chunks
    f32x8 s0 = fzero, s1 = fzero;
    s0 = wmma_bf16(qa0, load_b_frag(ktbase + (size_t)0  * SS + kv0 +  0, SS, lane), s0);
    s0 = wmma_bf16(qa1, load_b_frag(ktbase + (size_t)32 * SS + kv0 +  0, SS, lane), s0);
    s1 = wmma_bf16(qa0, load_b_frag(ktbase + (size_t)0  * SS + kv0 + 16, SS, lane), s1);
    s1 = wmma_bf16(qa1, load_b_frag(ktbase + (size_t)32 * SS + kv0 + 16, SS, lane), s1);

    // causal mask + online softmax (row stats reduced across 16-lane halves)
#pragma unroll
    for (int r = 0; r < 8; r++) {
      const int qrow = rbase + r;
      const int key  = kv0 + c0;
      float v0 = s0[r] * 0.125f; if (key      > qrow) v0 = -1e30f;
      float v1 = s1[r] * 0.125f; if (key + 16 > qrow) v1 = -1e30f;
      float mx = fmaxf(v0, v1);
#pragma unroll
      for (int off = 1; off < 16; off <<= 1) mx = fmaxf(mx, __shfl_xor(mx, off));
      const float mnew  = fmaxf(rmax[r], mx);
      const float alpha = __expf(rmax[r] - mnew);
      rmax[r] = mnew;
      const float p0 = __expf(v0 - mnew);
      const float p1 = __expf(v1 - mnew);
      float ps = p0 + p1;
#pragma unroll
      for (int off = 1; off < 16; off <<= 1) ps += __shfl_xor(ps, off);
      rsum[r] = rsum[r] * alpha + ps;
#pragma unroll
      for (int nj = 0; nj < 4; nj++) o[nj][r] *= alpha;
      // stage P tile (C-layout -> LDS row-major 16x32 bf16)
      const int prow = half * 8 + r;
      pw[prow * 32 + c0]      = (__bf16)p0;
      pw[prow * 32 + 16 + c0] = (__bf16)p1;
    }
    // wave-private LDS: wait for our own DS stores before cross-lane reload
    asm volatile("s_wait_dscnt 0" ::: "memory");

    // O += P @ V  (K = 32 keys, 4 n-frags over d = 64)
    bf16x16 pa = load_a_frag(pw, 32, lane);
    const __bf16* vrow = vbase + (size_t)kv0 * E3;
#pragma unroll
    for (int nj = 0; nj < 4; nj++)
      o[nj] = wmma_bf16(pa, load_b_frag(vrow + nj * 16, E3, lane), o[nj]);
  }

  // finalize: O / l, write ctx bf16 [B,S,E]
#pragma unroll
  for (int nj = 0; nj < 4; nj++)
#pragma unroll
    for (int r = 0; r < 8; r++) {
      const int qrow = rbase + r;
      float v = o[nj][r] / rsum[r];
      ctx[((size_t)(b * SS + qrow)) * EE + h * DD + nj * 16 + c0] = (__bf16)v;
    }
}

// ---------------------------------------------------------------------------
extern "C" void kernel_launch(void* const* d_in, const int* in_sizes, int n_in,
                              void* d_out, int out_size, void* d_ws, size_t ws_size,
                              hipStream_t stream) {
  (void)in_sizes; (void)n_in; (void)out_size; (void)ws_size;
  const float* x  = (const float*)d_in[0];
  const float* wi = (const float*)d_in[1];   // [3E, E]
  const float* bi = (const float*)d_in[2];   // [3E]
  const float* wo = (const float*)d_in[3];   // [E, E]
  const float* bo = (const float*)d_in[4];   // [E]

  char* ws = (char*)d_ws;
  size_t off = 0;
  auto take = [&](size_t bytes) -> char* {
    char* p = ws + off;
    off += (bytes + 255) & ~(size_t)255;
    return p;
  };

  __bf16* xb     = (__bf16*)take((size_t)MM * EE * 2);          // 8 MB (reused as ctx)
  __bf16* w_inT  = (__bf16*)take((size_t)EE * E3 * 2);          // 6 MB
  __bf16* w_outT = (__bf16*)take((size_t)EE * EE * 2);          // 2 MB
  __bf16* qkvb   = (__bf16*)take((size_t)MM * E3 * 2);          // 24 MB
  __bf16* ktb    = (__bf16*)take((size_t)BB * HH * DD * SS * 2);// 8 MB
  __bf16* ctxb   = xb;  // xb dead after GEMM1; reuse for attention output

  int n;
  n = MM * EE;
  cvt_f32_bf16<<<(n + 255) / 256, 256, 0, stream>>>(x, xb, n);
  n = E3 * EE;
  cvt_transpose<<<(n + 255) / 256, 256, 0, stream>>>(wi, w_inT, E3, EE);
  n = EE * EE;
  cvt_transpose<<<(n + 255) / 256, 256, 0, stream>>>(wo, w_outT, EE, EE);

  // qkv = x @ Win^T + bi   (M=4096, N=3072, K=1024), bf16 output
  gemm_nt<__bf16><<<dim3(E3 / 128, MM / 128), 256, 0, stream>>>(xb, w_inT, bi, qkvb, E3, EE);

  n = BB * HH * DD * SS;
  build_kt<<<(n + 255) / 256, 256, 0, stream>>>(qkvb, ktb);

  attn_kernel<<<dim3(SS / 64, HH, BB), 128, 0, stream>>>(qkvb, ktb, ctxb);

  // out = ctx @ Wout^T + bo  (M=4096, N=1024, K=1024), fp32 output
  gemm_nt<float><<<dim3(EE / 128, MM / 128), 256, 0, stream>>>(ctxb, w_outT, bo,
                                                               (float*)d_out, EE, EE);
}
